// MoELayer_87969520157162
// MI455X (gfx1250) — matrix operand order repaired and checked by reading, not compiled, not run on hardware
//
// MI455X (gfx1250) MoE layer: sparse top-2 dispatch + bf16 WMMA GEMMs.
//
// Roofline: sparse FLOPs ~275 GFLOP (vs 1.1 TFLOP dense reference); weight
// traffic 268 MB fp32 ~11.5us @ 23.3 TB/s -> compute bound -> bf16 WMMA
// (v_wmma_f32_16x16x32_bf16, fp32 accum) instead of the weak 16x16x4 f32 WMMA.
//
// v3 changes (from asm feedback):
//  - all fp32->bf16 conversion now uses hardware v_cvt_pk_bf16_f32 (1 VALU op
//    per 2 elements) instead of software RNE (~3 VALU ops/element) — the v2
//    histogram showed valu=3430 dominated by v_bfe_u32/v_add3_u32 rounding.
// Kept from v2: packed ds_store_b64 B fill, double-buffered LDS (1 barrier per
// K-step), async global->LDS DMA for gemm2's A tile (+s_wait_asynccnt).
//
// Workspace requirement: ~137 MB (bf16 intermediate h for padded slots).
#include <hip/hip_runtime.h>
#include <math.h>

#define NTOK   8192                 // B*S
#define DDIM   1024
#define FDIM   4096
#define NEXP   8
#define MAXP   (2*NTOK + NEXP*128)  // 17408 padded slots max
#define MAXT   (MAXP/128)           // 136 row tiles max

typedef __attribute__((ext_vector_type(16))) __bf16         bf16x16;
typedef __attribute__((ext_vector_type(8)))  float          f32x8;
typedef __attribute__((ext_vector_type(8)))  unsigned short u16x8;
typedef __attribute__((ext_vector_type(4)))  unsigned short u16x4;

union FragAB { bf16x16 bf; u16x8 us[2]; };
union PK8 { u16x8 us; unsigned u[4]; };
union PK4 { u16x4 us; unsigned u[2]; };

// Hardware packed convert: 2x f32 -> packed bf16 (RNE), one VALU op.
__device__ __forceinline__ unsigned pk2(float lo, float hi) {
  unsigned r;
  asm("v_cvt_pk_bf16_f32 %0, %1, %2" : "=v"(r) : "v"(lo), "v"(hi));
  return r;
}

__device__ __forceinline__ u16x8 cvt8(float4 a, float4 b) {
  PK8 r;
  r.u[0] = pk2(a.x, a.y); r.u[1] = pk2(a.z, a.w);
  r.u[2] = pk2(b.x, b.y); r.u[3] = pk2(b.z, b.w);
  return r.us;
}

// ---------------- 1. Router: one wave32 per token ----------------
__global__ __launch_bounds__(256) void moe_router(
    const float* __restrict__ x, const float* __restrict__ rw,
    const float* __restrict__ rb, int* __restrict__ counts,
    int2* __restrict__ tok_e, float2* __restrict__ tok_w)
{
  const int wave = threadIdx.x >> 5, lane = threadIdx.x & 31;
  const int t = blockIdx.x * 8 + wave;
  float acc[NEXP];
#pragma unroll
  for (int e = 0; e < NEXP; ++e) acc[e] = 0.f;
  const float* xr = x + (size_t)t * DDIM;
  for (int d = lane; d < DDIM; d += 32) {        // coalesced over lanes
    float xv = xr[d];
    float4 r0 = *(const float4*)(rw + d * NEXP);
    float4 r1 = *(const float4*)(rw + d * NEXP + 4);
    acc[0] += xv*r0.x; acc[1] += xv*r0.y; acc[2] += xv*r0.z; acc[3] += xv*r0.w;
    acc[4] += xv*r1.x; acc[5] += xv*r1.y; acc[6] += xv*r1.z; acc[7] += xv*r1.w;
  }
#pragma unroll
  for (int e = 0; e < NEXP; ++e) {
#pragma unroll
    for (int off = 16; off > 0; off >>= 1)
      acc[e] += __shfl_xor(acc[e], off, 32);
    acc[e] += rb[e];
  }
  // top-2 on logits; strict > keeps lowest index on ties (matches lax.top_k)
  int e0 = 0;
#pragma unroll
  for (int e = 1; e < NEXP; ++e) if (acc[e] > acc[e0]) e0 = e;
  int e1 = (e0 == 0) ? 1 : 0;
#pragma unroll
  for (int e = 0; e < NEXP; ++e) if (e != e0 && acc[e] > acc[e1]) e1 = e;
  float b  = expf(acc[e1] - acc[e0]);            // renormalized top-2 weights
  float w0 = 1.f / (1.f + b);
  if (lane == 0) {
    atomicAdd(&counts[e0], 1);
    atomicAdd(&counts[e1], 1);
    tok_e[t] = make_int2(e0, e1);
    tok_w[t] = make_float2(w0, 1.f - w0);
  }
}

// ---------------- 2. Padded prefix offsets (tiny, 1 thread) ----------------
__global__ void moe_offsets(const int* __restrict__ counts, int* __restrict__ offsets,
                            int* __restrict__ cursor, int* __restrict__ ntiles,
                            int* __restrict__ slot_token, float* __restrict__ slot_cw)
{
  int o = 0;
  offsets[0] = 0;
  for (int e = 0; e < NEXP; ++e) {
    int c = counts[e];
    int p = (c + 127) & ~127;                     // pad to M-tile
    for (int i = c; i < p; ++i) { slot_token[o + i] = -1; slot_cw[o + i] = 0.f; }
    cursor[e] = 0;
    o += p;
    offsets[e + 1] = o;
  }
  *ntiles = o >> 7;
}

// ---------------- 3. Fill slot lists (atomics pick position, not value) -----
__global__ __launch_bounds__(256) void moe_fill(
    const int2* __restrict__ tok_e, const float2* __restrict__ tok_w,
    const int* __restrict__ offsets, int* __restrict__ cursor,
    int* __restrict__ slot_token, float* __restrict__ slot_cw)
{
  int t = blockIdx.x * 256 + threadIdx.x;
  if (t >= NTOK) return;
  int2 e   = tok_e[t];
  float2 w = tok_w[t];
  int s0 = offsets[e.x] + atomicAdd(&cursor[e.x], 1);
  slot_token[s0] = t; slot_cw[s0] = w.x;
  int s1 = offsets[e.y] + atomicAdd(&cursor[e.y], 1);
  slot_token[s1] = t; slot_cw[s1] = w.y;
}

// ---- shared GEMM helpers ----------------------------------------------------
// B tile: thread covers 4 K-rows x 4 cols -> 4 coalesced float4 loads,
// 4 packed ds_store_b64 into col-major LDS [n][k].
__device__ __forceinline__ void load_b_regs(const float* __restrict__ W, int ldw,
                                            int k0, int col0, int kq, int cg,
                                            float (&bv)[4][4])
{
#pragma unroll
  for (int j = 0; j < 4; ++j)
    *(float4*)&bv[j][0] = *(const float4*)(W + (size_t)(k0 + kq + j) * ldw + col0 + cg);
}

__device__ __forceinline__ void store_b_lds(unsigned short (*__restrict__ B)[32],
                                            int kq, int cg, const float (&bv)[4][4])
{
#pragma unroll
  for (int c = 0; c < 4; ++c) {
    PK4 pk;
    pk.u[0] = pk2(bv[0][c], bv[1][c]);
    pk.u[1] = pk2(bv[2][c], bv[3][c]);
    *(u16x4*)&B[cg + c][kq] = pk.us;              // 8B aligned (kq % 4 == 0)
  }
}

// One 128x128x32 MMA step: A frag per wave strip, B frags preloaded 4 at a time.
__device__ __forceinline__ void mma_step(const unsigned short (*__restrict__ A)[32],
                                         const unsigned short (*__restrict__ B)[32],
                                         int wave, int lane, f32x8 (&acc)[8])
{
  FragAB fa;
  const int m  = wave * 16 + (lane & 15);
  const int kh = (lane >> 4) * 8;     // A: lanes 0-15 K{0..7,16..23}, 16-31 K{8..15,24..31}
  fa.us[0] = *(const u16x8*)&A[m][kh];
  fa.us[1] = *(const u16x8*)&A[m][kh + 16];
  const int n0 = lane & 15;
  const int kb = (lane >> 4) * 8;     // B: per-lane column, same K order as A
#pragma unroll
  for (int g = 0; g < 2; ++g) {
    FragAB fb[4];
#pragma unroll
    for (int i = 0; i < 4; ++i) {
      const int nt = g * 4 + i;
      fb[i].us[0] = *(const u16x8*)&B[nt * 16 + n0][kb];
      fb[i].us[1] = *(const u16x8*)&B[nt * 16 + n0][kb + 16];
    }
#pragma unroll
    for (int i = 0; i < 4; ++i)
      acc[g * 4 + i] = __builtin_amdgcn_wmma_f32_16x16x32_bf16(
          false, fa.bf, false, fb[i].bf, (short)0, acc[g * 4 + i], false, false);
  }
}

// ---------------- 4. GEMM1: h = gelu(gather(x) @ w1[e] + b1[e]) -------------
// 128x128 tile, 256 thr (8 waves), K-step 32, double-buffered LDS.
__global__ __launch_bounds__(256) void moe_gemm1(
    const float* __restrict__ x, const float* __restrict__ w1,
    const float* __restrict__ b1, const int* __restrict__ slot_token,
    const int* __restrict__ offsets, const int* __restrict__ ntiles,
    unsigned short* __restrict__ h)
{
  if ((int)blockIdx.x >= *ntiles) return;          // block-uniform early exit
  const int row0 = blockIdx.x * 128;
  const int col0 = blockIdx.y * 128;
  int e = 0;
#pragma unroll
  for (int i = 1; i < NEXP; ++i) if (row0 >= offsets[i]) e = i;
  const float* W = w1 + (size_t)e * DDIM * FDIM;

  const int tid  = threadIdx.x;
  const int arow = tid >> 1,        acol = (tid & 1) * 16;  // A: 2 thr/row
  const int kq   = (tid >> 5) * 4,  cg   = (tid & 31) * 4;  // B: 4Kx4N sub-tile
  const int wave = tid >> 5,        lane = tid & 31;
  const int tok  = slot_token[row0 + arow];
  const float* xr = (tok >= 0) ? (x + (size_t)tok * DDIM) : nullptr;

  __shared__ __align__(16) unsigned short Alds[2][128][32];  // row-major [m][k]
  __shared__ __align__(16) unsigned short Blds[2][128][32];  // col-major [n][k]

  f32x8 acc[8] = {};
  const int NIT = DDIM / 32;

  // prologue: stage tile 0
  {
    float4 a0 = {0,0,0,0}, a1 = a0, a2 = a0, a3 = a0;
    if (xr) {
      a0 = *(const float4*)(xr + acol);      a1 = *(const float4*)(xr + acol + 4);
      a2 = *(const float4*)(xr + acol + 8);  a3 = *(const float4*)(xr + acol + 12);
    }
    float bv[4][4];
    load_b_regs(W, FDIM, 0, col0, kq, cg, bv);
    *(u16x8*)&Alds[0][arow][acol]     = cvt8(a0, a1);
    *(u16x8*)&Alds[0][arow][acol + 8] = cvt8(a2, a3);
    store_b_lds(Blds[0], kq, cg, bv);
  }
  __syncthreads();

  for (int it = 0; it < NIT; ++it) {
    const int p    = it & 1;
    const bool more = (it + 1 < NIT);
    float4 a0, a1, a2, a3;
    float bv[4][4];
    if (more) {                                     // issue next-tile loads early
      const int k0 = (it + 1) * 32;
      if (xr) {
        a0 = *(const float4*)(xr + k0 + acol);      a1 = *(const float4*)(xr + k0 + acol + 4);
        a2 = *(const float4*)(xr + k0 + acol + 8);  a3 = *(const float4*)(xr + k0 + acol + 12);
      } else { a0 = {0,0,0,0}; a1 = a0; a2 = a0; a3 = a0; }
      load_b_regs(W, FDIM, k0, col0, kq, cg, bv);
      if (it + 2 < NIT)
        __builtin_prefetch(W + (size_t)((it + 2) * 32 + kq) * FDIM + col0 + cg, 0, 1);
    }
    mma_step(Alds[p], Blds[p], wave, lane, acc);    // compute current tile
    if (more) {                                     // fill other buffer
      *(u16x8*)&Alds[p ^ 1][arow][acol]     = cvt8(a0, a1);
      *(u16x8*)&Alds[p ^ 1][arow][acol + 8] = cvt8(a2, a3);
      store_b_lds(Blds[p ^ 1], kq, cg, bv);
    }
    __syncthreads();
  }

  // Epilogue: bias + exact GELU -> bf16 h. C layout: lanes 0-15 M=r, 16-31 M=r+8
  const int mrow = wave * 16 + ((lane >> 4) * 8);
  const int ncol = lane & 15;
#pragma unroll
  for (int nt = 0; nt < 8; ++nt) {
    const int col  = col0 + nt * 16 + ncol;
    const float bi = b1[e * FDIM + col];
    float g[8];
#pragma unroll
    for (int r = 0; r < 8; ++r) {
      float v = acc[nt][r] + bi;
      g[r] = 0.5f * v * (1.0f + erff(v * 0.70710678118654752f));
    }
#pragma unroll
    for (int r2 = 0; r2 < 4; ++r2) {                // packed cvt, split halves
      const unsigned pr = pk2(g[2 * r2], g[2 * r2 + 1]);
      h[(size_t)(row0 + mrow + 2 * r2)     * FDIM + col] = (unsigned short)(pr & 0xFFFFu);
      h[(size_t)(row0 + mrow + 2 * r2 + 1) * FDIM + col] = (unsigned short)(pr >> 16);
    }
  }
}

// ---------------- 5. GEMM2: out += cw * (h @ w2[e] + b2[e]) (scatter) -------
// A tile (already bf16) streamed with CDNA5 async global->LDS DMA.
__device__ __forceinline__ void async_a_tile(const unsigned short* __restrict__ gsrc,
                                             unsigned short* __restrict__ ldst)
{
  const unsigned l0 = (unsigned)(uintptr_t)ldst;   // low 32 bits = LDS offset
  const unsigned long long g0 = (unsigned long long)(uintptr_t)gsrc;
  asm volatile("global_load_async_to_lds_b128 %0, %1, off"
               :: "v"(l0), "v"(g0) : "memory");
  asm volatile("global_load_async_to_lds_b128 %0, %1, off offset:16"
               :: "v"(l0), "v"(g0) : "memory");
}

__global__ __launch_bounds__(256) void moe_gemm2(
    const unsigned short* __restrict__ h, const float* __restrict__ w2,
    const float* __restrict__ b2, const int* __restrict__ slot_token,
    const float* __restrict__ slot_cw, const int* __restrict__ offsets,
    const int* __restrict__ ntiles, float* __restrict__ out)
{
  if ((int)blockIdx.x >= *ntiles) return;
  const int row0 = blockIdx.x * 128;
  const int col0 = blockIdx.y * 128;
  int e = 0;
#pragma unroll
  for (int i = 1; i < NEXP; ++i) if (row0 >= offsets[i]) e = i;
  const float* W = w2 + (size_t)e * FDIM * DDIM;

  const int tid  = threadIdx.x;
  const int arow = tid >> 1,        acol = (tid & 1) * 16;
  const int kq   = (tid >> 5) * 4,  cg   = (tid & 31) * 4;
  const int wave = tid >> 5,        lane = tid & 31;
  const unsigned short* hr = h + (size_t)(row0 + arow) * FDIM;

  __shared__ __align__(16) unsigned short Alds[2][128][32];
  __shared__ __align__(16) unsigned short Blds[2][128][32];

  f32x8 acc[8] = {};
  const int NIT = FDIM / 32;

  // prologue: async A tile 0, B tile 0 through regs
  {
    async_a_tile(hr + acol, &Alds[0][arow][acol]);
    float bv[4][4];
    load_b_regs(W, DDIM, 0, col0, kq, cg, bv);
    store_b_lds(Blds[0], kq, cg, bv);
    asm volatile("s_wait_asynccnt 0x0" ::: "memory");
  }
  __syncthreads();

  for (int it = 0; it < NIT; ++it) {
    const int p    = it & 1;
    const bool more = (it + 1 < NIT);
    float bv[4][4];
    if (more) {
      const int k0 = (it + 1) * 32;
      async_a_tile(hr + k0 + acol, &Alds[p ^ 1][arow][acol]);  // DMA, no VGPRs
      load_b_regs(W, DDIM, k0, col0, kq, cg, bv);
      if (it + 2 < NIT)
        __builtin_prefetch(W + (size_t)((it + 2) * 32 + kq) * DDIM + col0 + cg, 0, 1);
    }
    mma_step(Alds[p], Blds[p], wave, lane, acc);
    if (more) {
      store_b_lds(Blds[p ^ 1], kq, cg, bv);
      asm volatile("s_wait_asynccnt 0x0" ::: "memory");        // drain A DMA
    }
    __syncthreads();
  }

  // Scatter-add: each out element gets exactly two fp32 adds (commutative ->
  // deterministic across replays) onto a zeroed out buffer.
  const int mrow = wave * 16 + ((lane >> 4) * 8);
  const int ncol = lane & 15;
#pragma unroll
  for (int r = 0; r < 8; ++r) {
    const int rg  = row0 + mrow + r;
    const int tok = slot_token[rg];
    if (tok < 0) continue;                           // padding row
    const float cw = slot_cw[rg];
#pragma unroll
    for (int nt = 0; nt < 8; ++nt) {
      const int col = col0 + nt * 16 + ncol;
      atomicAdd(&out[(size_t)tok * DDIM + col],
                cw * (acc[nt][r] + b2[e * DDIM + col]));
    }
  }
}

extern "C" void kernel_launch(void* const* d_in, const int* in_sizes, int n_in,
                              void* d_out, int out_size, void* d_ws, size_t ws_size,
                              hipStream_t stream)
{
  (void)in_sizes; (void)n_in; (void)ws_size;
  const float* x  = (const float*)d_in[0];
  const float* rw = (const float*)d_in[1];
  const float* rb = (const float*)d_in[2];
  const float* w1 = (const float*)d_in[3];
  const float* b1 = (const float*)d_in[4];
  const float* w2 = (const float*)d_in[5];
  const float* b2 = (const float*)d_in[6];
  float* out = (float*)d_out;

  // Workspace layout (total ~137 MB)
  char* ws = (char*)d_ws;
  int*    counts     = (int*)   (ws + 0);                       // 8
  int*    offsets    = (int*)   (ws + 64);                      // 9
  int*    cursor     = (int*)   (ws + 128);                     // 8
  int*    ntiles     = (int*)   (ws + 192);                     // 1
  int2*   tok_e      = (int2*)  (ws + 256);                     // NTOK
  float2* tok_w      = (float2*)(ws + 256 + NTOK*8);            // NTOK
  int*    slot_token = (int*)   (ws + 256 + NTOK*16);           // MAXP
  float*  slot_cw    = (float*) (ws + 256 + NTOK*16 + MAXP*4);  // MAXP
  unsigned short* h  = (unsigned short*)(ws + 256 + NTOK*16 + MAXP*8); // MAXP*FDIM bf16

  hipMemsetAsync(ws, 0, 256, stream);                           // counts/cursors
  hipMemsetAsync(d_out, 0, (size_t)out_size * sizeof(float), stream);

  moe_router <<<NTOK/8,   256, 0, stream>>>(x, rw, rb, counts, tok_e, tok_w);
  moe_offsets<<<1,        1,   0, stream>>>(counts, offsets, cursor, ntiles,
                                            slot_token, slot_cw);
  moe_fill   <<<NTOK/256, 256, 0, stream>>>(tok_e, tok_w, offsets, cursor,
                                            slot_token, slot_cw);
  moe_gemm1  <<<dim3(MAXT, FDIM/128), 256, 0, stream>>>(x, w1, b1, slot_token,
                                                        offsets, ntiles, h);
  moe_gemm2  <<<dim3(MAXT, DDIM/128), 256, 0, stream>>>(h, w2, b2, slot_token,
                                                        slot_cw, offsets, ntiles, out);
}
